// BinaryTreeLSTM_63453846831534
// MI455X (gfx1250) — matrix operand order repaired and checked by reading, not compiled, not run on hardware
//
#include <hip/hip_runtime.h>
#include <hip/hip_bf16.h>
#include <stdint.h>

typedef __attribute__((ext_vector_type(16))) __bf16       v16bf;
typedef __attribute__((ext_vector_type(8)))  float        v8f;
typedef __attribute__((ext_vector_type(4)))  unsigned int vu4;
typedef __attribute__((ext_vector_type(4)))  unsigned int u32x4;
typedef __attribute__((ext_vector_type(8)))  int          i32x8;
typedef __attribute__((ext_vector_type(4)))  int          i32x4;
typedef unsigned short u16;

static_assert(sizeof(v16bf) == 32, "v16bf must be 32 bytes");

#if defined(__has_builtin)
#if __has_builtin(__builtin_amdgcn_tensor_load_to_lds)
#define HAVE_TDM 1
#endif
#endif
#ifndef HAVE_TDM
#define HAVE_TDM 0
#endif

// therock-10.0 headers ship the TDM header and use the 6-arg builtin;
// ROCm 7.2 (clang-22) has the 5-arg form.
#if __has_include(<hip/amd_detail/amd_gfx1250_TDM.h>)
#define TDM_ARGS6 1
#else
#define TDM_ARGS6 0
#endif

// ---------------------------------------------------------------- helpers

__device__ __forceinline__ float sigmoidf_(float x) {
  return 1.0f / (1.0f + __expf(-x));
}

// Pack 8 consecutive f32 -> 8 bf16 and store as one b128 to LDS.
// Elements with (kg0+e >= kmax) are zero-padded (load suppressed).
__device__ __forceinline__ void store8bf(u16* dst, const float* src, int kg0, int kmax) {
  union { vu4 u; __bf16 h[8]; } pk;
#pragma unroll
  for (int e = 0; e < 8; ++e) {
    float f = (kg0 + e < kmax) ? src[e] : 0.f;
    pk.h[e] = (__bf16)f;
  }
  *reinterpret_cast<vu4*>(dst) = pk.u;
}

__device__ __forceinline__ void store8bf_nc(u16* dst, const float* src) {
  union { vu4 u; __bf16 h[8]; } pk;
#pragma unroll
  for (int e = 0; e < 8; ++e) pk.h[e] = (__bf16)src[e];
  *reinterpret_cast<vu4*>(dst) = pk.u;
}

// A fragment: 16x32 bf16 (MxK).  lane<16: rows M=lane, K = {0..7,16..23};
// lane>=16: same rows, K = {8..15,24..31}.  LDS row-major bf16, stride 64.
__device__ __forceinline__ v16bf load_a_frag(const u16* s, int rowTile, int lane, int kb) {
  int row  = rowTile + (lane & 15);
  int off  = kb + ((lane >> 4) << 3);
  union { vu4 u[2]; v16bf v; } f;
  f.u[0] = *reinterpret_cast<const vu4*>(s + row * 64 + off);
  f.u[1] = *reinterpret_cast<const vu4*>(s + row * 64 + off + 16);
  return f.v;
}

// B fragment: 32x16 bf16 (KxN), stored as W[col][k] row-major (stride 64):
// lane<16: col=lane, K = {0..7,16..23} ; lane>=16: col=lane-16, K = {8..15,24..31}.
__device__ __forceinline__ v16bf load_b_frag(const u16* s, int colTile, int lane, int kb) {
  int col = colTile + (lane & 15);
  int off = kb + ((lane >> 4) << 3);
  union { vu4 u[2]; v16bf v; } f;
  f.u[0] = *reinterpret_cast<const vu4*>(s + col * 64 + off);
  f.u[1] = *reinterpret_cast<const vu4*>(s + col * 64 + off + 16);
  return f.v;
}

__device__ __forceinline__ v8f wmma_bf16(v16bf a, v16bf b, v8f c) {
  return __builtin_amdgcn_wmma_f32_16x16x32_bf16(false, a, false, b, (short)0, c,
                                                 false, false);
}

// ---------------------------------------------------------------- leaf
// c = X @ cx_w^T + cx_b ; h = sigmoid(X @ ox_w^T + ox_b) * tanh(c)
// X: [R=131072, 300] f32 ; weights [256,300] f32.
// WG tile: 64 rows x 64 cols, 8 waves, each wave 32r x 16c for BOTH mats.
#define LEAF_IN 300

__global__ __launch_bounds__(256)
void treelstm_leaf(const float* __restrict__ x,
                   const float* __restrict__ cxw, const float* __restrict__ cxb,
                   const float* __restrict__ oxw, const float* __restrict__ oxb,
                   float* __restrict__ c_out, u16* __restrict__ h_out) {
  __shared__ __align__(16) u16 sA [64 * 64];
  __shared__ __align__(16) u16 sWc[64 * 64];
  __shared__ __align__(16) u16 sWo[64 * 64];

  const int tid  = threadIdx.x;
  const int lane = tid & 31;
  const int wave = tid >> 5;
  const int rowBase = blockIdx.x * 64;
  const int colBase = blockIdx.y * 64;
  const int wr = (wave >> 2) * 32;   // 0 / 32
  const int wc = (wave & 3) * 16;    // 0,16,32,48

  v8f aC0 = {}, aC1 = {}, aO0 = {}, aO1 = {};

  for (int kc = 0; kc < 5; ++kc) {          // Kpad = 320, chunks of 64
    const int k0 = kc * 64;
    // 8 halves per thread per iter -> 2 iters for each 64x64 tile
#pragma unroll
    for (int i8 = tid; i8 < (64 * 64) / 8; i8 += 256) {
      int r  = i8 >> 3;
      int kl = (i8 & 7) * 8;
      int kg = k0 + kl;
      store8bf(sA  + r * 64 + kl, x   + (size_t)(rowBase + r) * LEAF_IN + kg, kg, LEAF_IN);
      store8bf(sWc + r * 64 + kl, cxw + (size_t)(colBase + r) * LEAF_IN + kg, kg, LEAF_IN);
      store8bf(sWo + r * 64 + kl, oxw + (size_t)(colBase + r) * LEAF_IN + kg, kg, LEAF_IN);
    }
    if (kc + 1 < 5)
      __builtin_prefetch(x + (size_t)rowBase * LEAF_IN + k0 + 64, 0, 0);
    __syncthreads();
#pragma unroll
    for (int ks = 0; ks < 2; ++ks) {
      const int kb = ks * 32;
      v16bf a0 = load_a_frag(sA, wr,      lane, kb);
      v16bf a1 = load_a_frag(sA, wr + 16, lane, kb);
      v16bf bc = load_b_frag(sWc, wc, lane, kb);
      v16bf bo = load_b_frag(sWo, wc, lane, kb);
      aC0 = wmma_bf16(a0, bc, aC0);
      aC1 = wmma_bf16(a1, bc, aC1);
      aO0 = wmma_bf16(a0, bo, aO0);
      aO1 = wmma_bf16(a1, bo, aO1);
    }
    __syncthreads();
  }

  // epilogue: C/D layout -> lane<16: M=v ; lane>=16: M=v+8 ; N=lane&15
  const int col  = colBase + wc + (lane & 15);
  const float cb = cxb[col];
  const float ob = oxb[col];
  const int rOff = (lane >> 4) * 8;
#pragma unroll
  for (int v = 0; v < 8; ++v) {
    int r0 = rowBase + wr + rOff + v;
    float c0 = aC0[v] + cb;
    float o0 = sigmoidf_(aO0[v] + ob);
    c_out[(size_t)r0 * 256 + col] = c0;
    h_out[(size_t)r0 * 256 + col] = __builtin_bit_cast(u16, (__bf16)(o0 * tanhf(c0)));
    int r1 = r0 + 16;
    float c1 = aC1[v] + cb;
    float o1 = sigmoidf_(aO1[v] + ob);
    c_out[(size_t)r1 * 256 + col] = c1;
    h_out[(size_t)r1 * 256 + col] = __builtin_bit_cast(u16, (__bf16)(o1 * tanhf(c1)));
  }
}

// ---------------------------------------------------------------- compose
// A row r = concat(h[2r], h[2r+1]) -> 512 contiguous bf16 in h_prev.
// B col (g*256+o): k<256 -> lh_w[g][o][k], else rh_w[g][o][k-256].
// WG tile: 128 rows x 32 o-cols (x 4 gates). 8 waves: wave = 32r x 16c x 4g.
// A tile is a pure bf16 2D copy -> Tensor Data Mover (one D# per K-chunk).
__global__ __launch_bounds__(256)
void treelstm_compose(const float* __restrict__ c_prev,   // [2R,256] f32
                      const u16*   __restrict__ h_prev,   // [2R,256] bf16
                      const float* __restrict__ lhw, const float* __restrict__ lhb,
                      const float* __restrict__ rhw, const float* __restrict__ rhb,
                      float* __restrict__ c_out,          // [R,256] f32
                      u16*   __restrict__ h_out,          // [R,256] bf16
                      float* __restrict__ h_out_f32) {    // optional (last level)
  __shared__ __align__(16) u16 sA[128 * 64];   // rows x k
  __shared__ __align__(16) u16 sB[128 * 64];   // outcols(4g x 32o) x k

  const int tid  = threadIdx.x;
  const int lane = tid & 31;
  const int wave = tid >> 5;
  const int rowBase = blockIdx.x * 128;
  const int oBase   = blockIdx.y * 32;
  const int wr = (wave >> 1) * 32;   // 0,32,64,96
  const int wc = (wave & 1) * 16;    // 0,16

  v8f acc[2][4] = {};                // [row-tile][gate]

  for (int kc = 0; kc < 8; ++kc) {   // K = 512, chunks of 64
    const int k0 = kc * 64;

#if HAVE_TDM
    if (wave == 0) {
      // TDM: load 2D tile (tile_dim0=64 elems x tile_dim1=128 rows, 2B elems,
      // tensor row stride 512 elems) from h_prev into sA.  D# per ISA 8.3/8.4.
      const unsigned lds_lo = (unsigned)(uintptr_t)(void*)sA;
      const uint64_t ga = (uint64_t)(uintptr_t)(h_prev + (size_t)rowBase * 512 + k0);
      const unsigned TD = 1u << 20;  // generous tensor dims (no OOB clipping)
      u32x4 g0;
      g0[0] = 1u;                                             // count=1, user mode
      g0[1] = lds_lo;                                         // lds_addr
      g0[2] = (unsigned)ga;                                   // global_addr[31:0]
      g0[3] = (unsigned)((ga >> 32) & 0x01FFFFFFu) | (2u << 30); // [56:32] | type=2
      i32x8 g1;
      g1[0] = (int)(1u << 16);                                // data_size=1 (2B)
      g1[1] = (int)((TD & 0xFFFFu) << 16);                    // tensor_dim0 lo16
      g1[2] = (int)((TD >> 16) | ((TD & 0xFFFFu) << 16));     // td0 hi16 | td1 lo16
      g1[3] = (int)((TD >> 16) | (64u << 16));                // td1 hi16 | tile_dim0=64
      g1[4] = (int)128;                                       // tile_dim1=128
      g1[5] = (int)512;                                       // tensor_dim0_stride lo32
      g1[6] = 0;                                              // stride hi16 | td1_stride
      g1[7] = 0;
      const i32x4 z4 = {0, 0, 0, 0};
#if TDM_ARGS6
      const i32x8 z8 = {0, 0, 0, 0, 0, 0, 0, 0};
      __builtin_amdgcn_tensor_load_to_lds(g0, g1, z4, z4, z8, 0);
#else
      __builtin_amdgcn_tensor_load_to_lds(g0, g1, z4, z4, 0);
#endif
    }
#else
    // fallback: vectorized b128 copy (h_prev already bf16)
#pragma unroll
    for (int i8 = tid; i8 < (128 * 64) / 8; i8 += 256) {
      int r = i8 >> 3, kl = (i8 & 7) * 8;
      *reinterpret_cast<vu4*>(sA + r * 64 + kl) =
          *reinterpret_cast<const vu4*>(h_prev + (size_t)(rowBase + r) * 512 + k0 + kl);
    }
#endif

    // B tile: chunk never straddles k=256, so source matrix is uniform.
    const float* w  = (k0 < 256) ? lhw : rhw;
    const int   kb0 = k0 & 255;
#pragma unroll
    for (int i8 = tid; i8 < (128 * 64) / 8; i8 += 256) {  // 4 iters
      int r  = i8 >> 3;              // outcol local: g*32 + j
      int kl = (i8 & 7) * 8;
      int g = r >> 5, j = r & 31;
      store8bf_nc(sB + r * 64 + kl,
                  w + ((size_t)g * 256 + (oBase + j)) * 256 + kb0 + kl);
    }
    if (kc + 1 < 8)
      __builtin_prefetch(h_prev + (size_t)rowBase * 512 + k0 + 64, 0, 0);

#if HAVE_TDM
    if (wave == 0) __builtin_amdgcn_s_wait_tensorcnt(0);
#endif
    __syncthreads();
#pragma unroll
    for (int ks = 0; ks < 2; ++ks) {
      const int kb = ks * 32;
      v16bf a0 = load_a_frag(sA, wr,      lane, kb);
      v16bf a1 = load_a_frag(sA, wr + 16, lane, kb);
#pragma unroll
      for (int g = 0; g < 4; ++g) {
        v16bf b = load_b_frag(sB, g * 32 + wc, lane, kb);
        acc[0][g] = wmma_bf16(a0, b, acc[0][g]);
        acc[1][g] = wmma_bf16(a1, b, acc[1][g]);
      }
    }
    __syncthreads();
  }

  // epilogue: gates + cell update
  const int o = oBase + wc + (lane & 15);
  const float b0 = lhb[0 * 256 + o] + rhb[0 * 256 + o];
  const float b1 = lhb[1 * 256 + o] + rhb[1 * 256 + o];
  const float b2 = lhb[2 * 256 + o] + rhb[2 * 256 + o];
  const float b3 = lhb[3 * 256 + o] + rhb[3 * 256 + o];
  const int rOff = (lane >> 4) * 8;
#pragma unroll
  for (int rt = 0; rt < 2; ++rt) {
#pragma unroll
    for (int v = 0; v < 8; ++v) {
      int r = rowBase + wr + rt * 16 + rOff + v;
      float gi = sigmoidf_(acc[rt][0][v] + b0);
      float lf = sigmoidf_(acc[rt][1][v] + b1);
      float rf = sigmoidf_(acc[rt][2][v] + b2);
      float gu = tanhf    (acc[rt][3][v] + b3);
      float lc = c_prev[(size_t)(2 * r)     * 256 + o];
      float rc = c_prev[(size_t)(2 * r + 1) * 256 + o];
      float cn = gi * gu + lf * lc + rf * rc;
      float hn = tanhf(cn);
      c_out[(size_t)r * 256 + o] = cn;
      h_out[(size_t)r * 256 + o] = __builtin_bit_cast(u16, (__bf16)hn);
      if (h_out_f32) h_out_f32[(size_t)r * 256 + o] = hn;
    }
  }
}

// ---------------------------------------------------------------- launch

extern "C" void kernel_launch(void* const* d_in, const int* in_sizes, int n_in,
                              void* d_out, int out_size, void* d_ws, size_t ws_size,
                              hipStream_t stream) {
  const float* embs = (const float*)d_in[0];
  const float* cx_w = (const float*)d_in[1];
  const float* cx_b = (const float*)d_in[2];
  const float* ox_w = (const float*)d_in[3];
  const float* ox_b = (const float*)d_in[4];
  const float* lh_w = (const float*)d_in[5];
  const float* lh_b = (const float*)d_in[6];
  const float* rh_w = (const float*)d_in[7];
  const float* rh_b = (const float*)d_in[8];
  float* out = (float*)d_out;

  const size_t B = 512, L = 256, M = 256;
  const size_t R0 = B * L;                 // 131072 leaf rows

  // workspace layout (ping-pong): c buffers f32, h buffers bf16
  char* ws = (char*)d_ws;
  float* cP = (float*)ws;                                  // R0 * M f32
  u16*   hP = (u16*)  (ws + R0 * M * 4);                   // R0 * M bf16
  float* cQ = (float*)(ws + R0 * M * 6);                   // R0/2 * M f32
  u16*   hQ = (u16*)  (ws + R0 * M * 6 + (R0 / 2) * M * 4);

  // leaf: 131072 x {300 -> 256} dual GEMM
  dim3 gridL((unsigned)(R0 / 64), (unsigned)(M / 64));
  treelstm_leaf<<<gridL, 256, 0, stream>>>(embs, cx_w, cx_b, ox_w, ox_b, cP, hP);

  const float* cin = cP;
  const u16*   hin = hP;
  float* cAlt[2] = {cQ, cP};
  u16*   hAlt[2] = {hQ, hP};

  int nodes = (int)L;
  for (int lvl = 0; lvl < 8; ++lvl) {
    nodes >>= 1;
    size_t R = B * (size_t)nodes;          // 65536 ... 512
    float* co; u16* ho; float* hf32 = nullptr;
    if (nodes == 1) { co = out; ho = hQ; hf32 = out + B * M; }
    else           { co = cAlt[lvl & 1]; ho = hAlt[lvl & 1]; }
    dim3 grid((unsigned)(R / 128), (unsigned)(M / 32));
    treelstm_compose<<<grid, 256, 0, stream>>>(cin, hin, lh_w, lh_b, rh_w, rh_b,
                                               co, ho, hf32);
    cin = co; hin = ho;
  }
}
